// GraphConvolutionNetwork_74448963108867
// MI455X (gfx1250) — matrix operand order repaired and checked by reading, not compiled, not run on hardware
//
#include <hip/hip_runtime.h>
#include <stdint.h>

// ---------------- types (raw ext-vectors to avoid HIP vector-struct ctor issues) --
typedef __attribute__((ext_vector_type(16))) __bf16          v16bf;
typedef __attribute__((ext_vector_type(16))) unsigned short  v16u;
typedef __attribute__((ext_vector_type(8)))  float           v8f;
typedef __attribute__((ext_vector_type(4)))  float           v4f;
typedef __attribute__((ext_vector_type(4)))  unsigned int    v4u;

union Frag {
    v16u u;      // 16 x u16 (raw bf16 bits)
    v16bf bf;    // view for the WMMA builtin
    v4u  q[2];   // two 16B loads
};

struct FragSet { Frag a, b0, b1, b2, b3; };

#define GCN_N 50000
#define GCN_E 800000

// ---------------- degree / normalization ------------------------------------------
__global__ void deg_init(float* deg, int n) {
    int i = blockIdx.x * blockDim.x + threadIdx.x;
    if (i < n) deg[i] = 1.0f;                       // self-loop
}
__global__ void deg_accum(float* deg, const int* __restrict__ dst, int e) {
    int i = blockIdx.x * blockDim.x + threadIdx.x;
    if (i < e) atomicAdd(&deg[dst[i]], 1.0f);
}
__global__ void deg_finish(float* dis, int n) {
    int i = blockIdx.x * blockDim.x + threadIdx.x;
    if (i < n) dis[i] = rsqrtf(dis[i]);             // deg >= 1 always
}
__global__ void edge_norm(float* __restrict__ nrm, const float* __restrict__ dis,
                          const int* __restrict__ src, const int* __restrict__ dst, int e) {
    int i = blockIdx.x * blockDim.x + threadIdx.x;
    if (i < e) nrm[i] = dis[src[i]] * dis[dst[i]];
}

// ---------------- aggregation: self-loop init (+ optional bias) -------------------
// Y[i][c] = X[i][c] * dis[i]^2 (+ bias[c])
__global__ void self_init(float* __restrict__ Y, const float* __restrict__ X,
                          const float* __restrict__ dis, const float* __restrict__ bias,
                          int c4, int n) {
    long long gid = (long long)blockIdx.x * blockDim.x + threadIdx.x;
    long long total = (long long)n * c4;
    if (gid >= total) return;
    int i = (int)(gid / c4);
    int c = (int)(gid % c4);
    float sn = dis[i] * dis[i];
    v4f v = ((const v4f*)X)[(size_t)i * c4 + c];
    v4f o = v * sn;
    if (bias) o += ((const v4f*)bias)[c];
    ((v4f*)Y)[(size_t)i * c4 + c] = o;
}

// ---------------- aggregation: edge scatter (L2-resident atomics) -----------------
__global__ void edge_scatter(float* __restrict__ Y, const float* __restrict__ X,
                             const int* __restrict__ src, const int* __restrict__ dst,
                             const float* __restrict__ nrm, int c4, int e) {
    long long gid = (long long)blockIdx.x * blockDim.x + threadIdx.x;
    long long total = (long long)e * c4;
    if (gid >= total) return;
    int ei = (int)(gid / c4);
    int c  = (int)(gid % c4);
    int s = src[ei], d = dst[ei];
    float w = nrm[ei];
    v4f v = ((const v4f*)X)[(size_t)s * c4 + c];
    float* yp = Y + ((size_t)d * c4 + c) * 4;
    atomicAdd(yp + 0, v.x * w);
    atomicAdd(yp + 1, v.y * w);
    atomicAdd(yp + 2, v.z * w);
    atomicAdd(yp + 3, v.w * w);
}

// ---------------- pack W (f32 row-major) into WMMA B-fragment bf16 layout ---------
// Wp[((ks*nCT + ct)*32 + lane)*16 + i] = bf16( W[ks*32 + 16*(lane>=16) + i][ct*16 + (lane&15)] )
__global__ void pack_w(const float* __restrict__ W, unsigned short* __restrict__ Wp,
                       int K, int outc) {
    int nCT = outc >> 4;
    int total = (K >> 5) * nCT * 32;
    int gid = blockIdx.x * blockDim.x + threadIdx.x;
    if (gid >= total) return;
    int lane = gid & 31;
    int t = gid >> 5;
    int ct = t % nCT;
    int ks = t / nCT;
    int kb  = ks * 32 + ((lane < 16) ? 0 : 16);
    int col = ct * 16 + (lane & 15);
    unsigned short* o = Wp + (size_t)gid * 16;
#pragma unroll
    for (int i = 0; i < 16; ++i)
        o[i] = (unsigned short)(__float_as_uint(W[(size_t)(kb + i) * outc + col]) >> 16);
}

// ---------------- pack X (f32 row-major) into WMMA A-fragment bf16 layout ---------
// A 16x32 bf16 fragment (ISA 7.12.2): lane holds row (lane&15), halves of wave split K.
__global__ void pack_x(const float* __restrict__ X, unsigned short* __restrict__ Xp,
                       int nRowTiles, int K) {
    int ksteps = K >> 5;
    int total = nRowTiles * ksteps * 32;
    int gid = blockIdx.x * blockDim.x + threadIdx.x;
    if (gid >= total) return;
    int lane = gid & 31;
    int t = gid >> 5;
    int ks = t % ksteps;
    int rt = t / ksteps;
    int row = rt * 16 + (lane & 15);
    int kb0 = ks * 32 + ((lane < 16) ? 0 : 16);
    const float* xr = X + (size_t)row * K + kb0;
    unsigned short* o = Xp + (size_t)gid * 16;
#pragma unroll
    for (int i = 0; i < 8; ++i) {
        o[i]     = (unsigned short)(__float_as_uint(xr[i])      >> 16);
        o[8 + i] = (unsigned short)(__float_as_uint(xr[8 + i])  >> 16);
    }
}

// ---------------- WMMA GEMM helpers -----------------------------------------------
__device__ __forceinline__ void load_set(FragSet& s, const v4u* ap, const v4u* bp) {
    s.a.q[0]  = ap[0];   s.a.q[1]  = ap[1];
    s.b0.q[0] = bp[0];   s.b0.q[1] = bp[1];
    s.b1.q[0] = bp[64];  s.b1.q[1] = bp[65];    // next 16-col tile: +32 lanes * 2
    s.b2.q[0] = bp[128]; s.b2.q[1] = bp[129];
    s.b3.q[0] = bp[192]; s.b3.q[1] = bp[193];
}
__device__ __forceinline__ void mma_set(const FragSet& s, v8f& a0, v8f& a1, v8f& a2, v8f& a3) {
    a0 = __builtin_amdgcn_wmma_f32_16x16x32_bf16(false, s.a.bf, false, s.b0.bf, (short)0, a0, false, false);
    a1 = __builtin_amdgcn_wmma_f32_16x16x32_bf16(false, s.a.bf, false, s.b1.bf, (short)0, a1, false, false);
    a2 = __builtin_amdgcn_wmma_f32_16x16x32_bf16(false, s.a.bf, false, s.b2.bf, (short)0, a2, false, false);
    a3 = __builtin_amdgcn_wmma_f32_16x16x32_bf16(false, s.a.bf, false, s.b3.bf, (short)0, a3, false, false);
}

// ---------------- WMMA GEMM: Out = relu?( Xp @ Wp + bias ) ------------------------
// Both operands pre-packed bf16 in fragment layout. Each wave: 16 rows x 64 cols
// (4 accumulators). Software-pipelined ping-pong fragment sets: loads for K-step
// k+1 are in flight while the WMMAs for K-step k execute (ksteps is even: 4 or 8).
__global__ void __launch_bounds__(256)
gemm_wmma(const unsigned short* __restrict__ Xp, const unsigned short* __restrict__ Wp,
          const float* __restrict__ bias, float* __restrict__ Out,
          int nRowTiles, int ksteps, int outc, int nColGroups, int relu) {
    int wave = blockIdx.x * (blockDim.x >> 5) + (threadIdx.x >> 5);
    int lane = threadIdx.x & 31;
    int rowtile = wave / nColGroups;
    int cgroup  = wave % nColGroups;
    if (rowtile >= nRowTiles) return;          // wave-uniform: EXEC all-ones inside

    int rowbase = rowtile * 16;
    int colbase = cgroup * 64;
    int nCT16 = outc >> 4;
    int laneLo = lane & 15;
    int hi = lane >> 4;

    v8f acc0 = {}, acc1 = {}, acc2 = {}, acc3 = {};
    const v4u* ap = (const v4u*)Xp + ((size_t)rowtile * ksteps * 32 + lane) * 2;
    const v4u* bp = (const v4u*)Wp + ((size_t)(colbase >> 4) * 32 + lane) * 2;
    const size_t aStep = 32 * 2;                 // one A fragment per K-step
    const size_t bStep = (size_t)nCT16 * 32 * 2; // row of 16-col tiles per K-step

    FragSet s0, s1;
    load_set(s0, ap, bp);                        // prologue: K-step 0
    for (int ks = 0; ks < ksteps; ks += 2) {
        load_set(s1, ap + aStep, bp + bStep);    // prefetch K-step ks+1
        mma_set(s0, acc0, acc1, acc2, acc3);     // consume ks (waits only its loads)
        ap += 2 * aStep;
        bp += 2 * bStep;
        if (ks + 2 < ksteps)
            load_set(s0, ap, bp);                // prefetch K-step ks+2
        mma_set(s1, acc0, acc1, acc2, acc3);     // consume ks+1
    }

    float bv0 = 0.f, bv1 = 0.f, bv2 = 0.f, bv3 = 0.f;
    if (bias) {
        bv0 = bias[colbase +  0 + laneLo];
        bv1 = bias[colbase + 16 + laneLo];
        bv2 = bias[colbase + 32 + laneLo];
        bv3 = bias[colbase + 48 + laneLo];
    }
#pragma unroll
    for (int r = 0; r < 8; ++r) {
        int orow = rowbase + r + hi * 8;
        float* op = Out + (size_t)orow * outc + colbase + laneLo;
        float o0 = acc0[r] + bv0, o1 = acc1[r] + bv1, o2 = acc2[r] + bv2, o3 = acc3[r] + bv3;
        if (relu) { o0 = fmaxf(o0, 0.f); o1 = fmaxf(o1, 0.f); o2 = fmaxf(o2, 0.f); o3 = fmaxf(o3, 0.f); }
        op[0] = o0; op[16] = o1; op[32] = o2; op[48] = o3;
    }
}

// ---------------- host orchestration ----------------------------------------------
extern "C" void kernel_launch(void* const* d_in, const int* in_sizes, int n_in,
                              void* d_out, int out_size, void* d_ws, size_t ws_size,
                              hipStream_t stream) {
    const int N = GCN_N, E = GCN_E;
    const float* x    = (const float*)d_in[0];
    const int*   ei   = (const int*)d_in[1];       // [2,E] int32
    const int*   srcI = ei;
    const int*   dstI = ei + E;
    const float* W[4] = { (const float*)d_in[3], (const float*)d_in[5],
                          (const float*)d_in[7], (const float*)d_in[9] };
    const float* B[4] = { (const float*)d_in[4], (const float*)d_in[6],
                          (const float*)d_in[8], (const float*)d_in[10] };
    float* out = (float*)d_out;

    // workspace carve (all regions fully initialized before use each call)
    uintptr_t p = (uintptr_t)d_ws;
    auto carve = [&](size_t bytes) -> void* {
        uintptr_t r = (p + 255) & ~(uintptr_t)255;
        p = r + bytes;
        return (void*)r;
    };
    float*          dis  = (float*)carve((size_t)N * 4);
    float*          nrm  = (float*)carve((size_t)E * 4);
    float*          bufA = (float*)carve((size_t)N * 256 * 4);
    float*          bufB = (float*)carve((size_t)N * 256 * 4);
    unsigned short* Wp   = (unsigned short*)carve((size_t)256 * 256 * 2);
    unsigned short* Xp   = (unsigned short*)carve((size_t)N * 256 * 2);

    const int T = 256;
    auto nb = [&](long long total) { return (unsigned)((total + T - 1) / T); };

    // --- normalization (once) ---
    deg_init  <<<nb(N), T, 0, stream>>>(dis, N);
    deg_accum <<<nb(E), T, 0, stream>>>(dis, dstI, E);
    deg_finish<<<nb(N), T, 0, stream>>>(dis, N);
    edge_norm <<<nb(E), T, 0, stream>>>(nrm, dis, srcI, dstI, E);

    auto launch_agg = [&](float* Y, const float* Xf, const float* bias, int C) {
        int c4 = C >> 2;
        self_init   <<<nb((long long)N * c4), T, 0, stream>>>(Y, Xf, dis, bias, c4, N);
        edge_scatter<<<nb((long long)E * c4), T, 0, stream>>>(Y, Xf, srcI, dstI, nrm, c4, E);
    };
    auto launch_gemm = [&](const float* Xin, const float* Wf, const float* bias,
                           float* O, int K, int outc, int relu) {
        const int rowTiles = N / 16;                // 50000/16 = 3125 exact
        const int ksteps = K >> 5;                  // 4 or 8 (even)
        int packWT = ksteps * (outc >> 4) * 32;
        pack_w<<<nb(packWT), T, 0, stream>>>(Wf, Wp, K, outc);
        long long packXT = (long long)rowTiles * ksteps * 32;
        pack_x<<<nb(packXT), T, 0, stream>>>(Xin, Xp, rowTiles, K);
        int nCG = outc / 64;
        long long waves = (long long)rowTiles * nCG;
        gemm_wmma<<<(unsigned)((waves + 7) / 8), 256, 0, stream>>>(
            Xp, Wp, bias, O, rowTiles, ksteps, outc, nCG, relu);
    };

    // Layer 0: aggregate in 128-dim (cheaper), then GEMM 128->256 (+bias,+relu)
    launch_agg (bufB, x, nullptr, 128);
    launch_gemm(bufB, W[0], B[0], bufA, 128, 256, 1);
    // Layer 1: aggregate 256, GEMM 256->256
    launch_agg (bufB, bufA, nullptr, 256);
    launch_gemm(bufB, W[1], B[1], bufA, 256, 256, 1);
    // Layer 2
    launch_agg (bufB, bufA, nullptr, 256);
    launch_gemm(bufB, W[2], B[2], bufA, 256, 256, 1);
    // Layer 3: GEMM 256->128 first (no bias/relu), then aggregate in 128-dim,
    //          folding bias b3 into the self-loop init. out = A*(x3 W3) + b3.
    launch_gemm(bufA, W[3], nullptr, bufB, 256, 128, 0);
    launch_agg (out, bufB, B[3], 128);
}
